// Node2Vec_88476326297730
// MI455X (gfx1250) — matrix operand order repaired and checked by reading, not compiled, not run on hardware
//
#include <hip/hip_runtime.h>
#include <math.h>

// ---- problem constants (from reference) ----
#define HD     256
#define FEATD  512
#define NN     32768   // BS * N_NODES
#define EE     32768   // edges
#define BSZ    32
#define NNODES 1024
#define VOC    780
#define EFK    514     // edge-mlp logical input width: 2H + 2
#define EFKS   516     // padded activation row stride (16B-aligned rows)
#define EFKP   544     // transposed-weight padded K stride (mult of 32)
#define CRANGE 30.0f

typedef __attribute__((ext_vector_type(16))) __bf16 bf16x16;
typedef __attribute__((ext_vector_type(8)))  float  f32x8;

union FragU { bf16x16 v; unsigned u[8]; };

__device__ __forceinline__ float siluf(float x) { return x / (1.0f + __expf(-x)); }

#define WAIT_ASYNC() asm volatile("s_wait_asynccnt 0x0" ::: "memory")

// =====================================================================
// WMMA GEMM: C[M,N] = epi(A[M,K]f32 * Bt + bias) (+res) (*mask[m])
//   A : f32 row-major [M x K], lda mult of 4 (16B rows), M % 128 == 0
//   Bt: bf16 TRANSPOSED weights [N x KP], KP = roundup(K,32), zero-padded;
//       fetched with GLOBAL_LOAD_ASYNC_TO_LDS_B128 (ASYNCcnt), double-buffered
//   N % 64 == 0. Only A needs a predicated K tail.
// block = 256 threads = 8 waves; tile 128(M) x 64(N); BK = 32
// =====================================================================
#define BM 128
#define BN 64
#define BK 32
#define ALD 36   // bf16 row stride: 72B rows -> 8B-aligned b64 stores, conflict-free
#define BLD 40   // bf16 row stride: 80B rows -> 16B-aligned b128 async writes, conflict-free

__global__ __launch_bounds__(256)
void gemm_bf16_wmma(const float* __restrict__ A, const __bf16* __restrict__ Bt,
                    const float* __restrict__ bias, const float* __restrict__ res,
                    const float* __restrict__ maskrow, float* __restrict__ C,
                    int M, int N, int K, int lda, int ldb /*=KP*/, int ldc, int act)
{
    __shared__ __align__(16) __bf16 As[BM * ALD];
    __shared__ __align__(16) __bf16 Bs[2][BN * BLD];

    const int t    = threadIdx.x;
    const int m0   = blockIdx.y * BM;
    const int n0   = blockIdx.x * BN;
    const int w    = t >> 5;        // wave id 0..7
    const int lane = t & 31;
    const int ln   = lane & 15;
    const int half = lane >> 4;
    const int waveM = w & 3;        // 4 waves across M
    const int waveN = w >> 2;       // 2 waves across N

    // A staging geometry: 128x32 f32 as 1024 float4-chunks (4/thread)
    int aRow[4], aCol[4];
    #pragma unroll
    for (int i = 0; i < 4; ++i) { int ch = i*256 + t; aRow[i] = ch >> 3; aCol[i] = (ch & 7) * 4; }
    // B async geometry: 64 rows x 64B, one b128 (16B) chunk per thread
    const int bN  = t >> 2;          // 0..63
    const int bCo = (t & 3) * 8;     // bf16 element offset (8 elems = 16B)

    float4 ra[4];
    f32x8 acc[2][2] = {};

    const int kFull  = K & ~(BK - 1);
    const int nFull  = kFull >> 5;
    const bool hasTail = (kFull < K);

    auto loadA = [&](int k0) {
        #pragma unroll
        for (int i = 0; i < 4; ++i)
            ra[i] = *(const float4*)&A[(long)(m0 + aRow[i]) * lda + k0 + aCol[i]];
    };

    auto stageA = [&]() {
        #pragma unroll
        for (int i = 0; i < 4; ++i) {
            union { __bf16 h[4]; unsigned long long u; } pk;
            pk.h[0] = (__bf16)ra[i].x; pk.h[1] = (__bf16)ra[i].y;
            pk.h[2] = (__bf16)ra[i].z; pk.h[3] = (__bf16)ra[i].w;
            *(unsigned long long*)&As[aRow[i] * ALD + aCol[i]] = pk.u;
        }
    };

    // one async b128 per lane: Bt[n0+bN][k0+bCo .. +7] -> Bs[buf][bN][bCo..]
    auto asyncB = [&](int k0, int buf) {
        unsigned lds = (unsigned)(unsigned long long)(const void*)&Bs[buf][bN * BLD + bCo];
        const __bf16* src = &Bt[(long)(n0 + bN) * ldb + k0 + bCo];
        asm volatile("global_load_async_to_lds_b128 %0, %1, off"
                     :: "v"(lds), "v"(src) : "memory");
    };

    auto mmaStep = [&](int buf) {
        // fragments per ISA 7.12.2 (wave32: A 16x32 bf16, B 32x16 bf16)
        FragU a[2], b[2];
        #pragma unroll
        for (int tm = 0; tm < 2; ++tm) {
            int mr = waveM * 32 + tm * 16 + ln;
            #pragma unroll
            for (int i = 0; i < 8; ++i) {
                int kk = ((i >= 4) ? 16 : 0) + half * 8 + (i & 3) * 2;
                a[tm].u[i] = *(const unsigned*)&As[mr * ALD + kk];
            }
        }
        #pragma unroll
        for (int tn = 0; tn < 2; ++tn) {
            int nc = waveN * 32 + tn * 16 + ln;
            #pragma unroll
            for (int i = 0; i < 8; ++i) {
                int kk = ((i >= 4) ? 16 : 0) + half * 8 + (i & 3) * 2;
                b[tn].u[i] = *(const unsigned*)&Bs[buf][nc * BLD + kk];
            }
        }
        #pragma unroll
        for (int tm = 0; tm < 2; ++tm)
            #pragma unroll
            for (int tn = 0; tn < 2; ++tn)
                acc[tm][tn] = __builtin_amdgcn_wmma_f32_16x16x32_bf16(
                    false, a[tm].v, false, b[tn].v, (short)0, acc[tm][tn],
                    false, false);
    };

    // ---- software-pipelined main loop over full K tiles ----
    if (nFull > 0) { asyncB(0, 0); loadA(0); }
    else if (hasTail) { asyncB(0, 0); }

    for (int j = 0; j < nFull; ++j) {
        int k0 = j << 5;
        WAIT_ASYNC();                 // B tile for this iter landed in LDS
        __syncthreads();              // all waves done reading As/Bs of prev iter
        stageA();                     // one s_wait_loadcnt, amortized per tile
        if (j + 1 < nFull) {
            if (k0 + 2 * BK < K)      // speculative prefetch two tiles ahead
                __builtin_prefetch(&A[(long)(m0 + aRow[0]) * lda + k0 + 2 * BK], 0, 1);
            asyncB(k0 + BK, (j + 1) & 1);
            loadA(k0 + BK);           // overlap global_load_b128 with ds + wmma
        } else if (hasTail) {
            asyncB(kFull, (j + 1) & 1);
        }
        __syncthreads();
        mmaStep(j & 1);
    }

    // ---- predicated A tail (only edge MLP: K=514 -> 2 extra; B pad is zero) ----
    if (hasTail) {
        WAIT_ASYNC();
        __syncthreads();
        #pragma unroll
        for (int i = 0; i < 16; ++i) {
            int idx = i * 256 + t;
            int r = idx >> 5, c = idx & 31;
            int gk = kFull + c;
            float v = (gk < K) ? A[(long)(m0 + r) * lda + gk] : 0.0f;
            As[r * ALD + c] = (__bf16)v;
        }
        __syncthreads();
        mmaStep(nFull & 1);
    }

    // ---- epilogue: +bias -> act -> +res -> *mask[m] ----
    if (m0 + BM <= M && n0 + BN <= N) {
        #pragma unroll
        for (int tm = 0; tm < 2; ++tm) {
            #pragma unroll
            for (int tn = 0; tn < 2; ++tn) {
                int n = n0 + waveN * 32 + tn * 16 + ln;
                float bv = bias ? bias[n] : 0.0f;
                #pragma unroll
                for (int r = 0; r < 8; ++r) {
                    int m = m0 + waveM * 32 + tm * 16 + half * 8 + r;
                    float v = acc[tm][tn][r] + bv;
                    if (act)     v  = siluf(v);
                    if (res)     v += res[(long)m * ldc + n];
                    if (maskrow) v *= maskrow[m];
                    C[(long)m * ldc + n] = v;
                }
            }
        }
    } else {
        #pragma unroll
        for (int tm = 0; tm < 2; ++tm) {
            #pragma unroll
            for (int tn = 0; tn < 2; ++tn) {
                int n = n0 + waveN * 32 + tn * 16 + ln;
                #pragma unroll
                for (int r = 0; r < 8; ++r) {
                    int m = m0 + waveM * 32 + tm * 16 + half * 8 + r;
                    if (m < M && n < N) {
                        float v = acc[tm][tn][r];
                        if (bias)    v += bias[n];
                        if (act)     v  = siluf(v);
                        if (res)     v += res[(long)m * ldc + n];
                        if (maskrow) v *= maskrow[m];
                        C[(long)m * ldc + n] = v;
                    }
                }
            }
        }
    }
}

// ---- weight convert+transpose: W[l][K][N] f32 -> Wt[l][N][KP] bf16, zero-padded ----
__global__ void wtrans(const float* __restrict__ W, __bf16* __restrict__ Wt,
                       int K, int N, int KP)
{
    int l = blockIdx.z;
    int n = blockIdx.y;
    int k = blockIdx.x * 256 + threadIdx.x;
    if (k >= KP) return;
    __bf16 v = (k < K) ? (__bf16)W[(long)l * K * N + (long)k * N + n] : (__bf16)0.0f;
    Wt[(long)l * N * KP + (long)n * KP + k] = v;
}

// ---- comb[n] = [ v_emb[v[n]] | fe[n] | size_emb[size[n]] ] ----
__global__ void build_comb(const int* __restrict__ v, const int* __restrict__ sz,
                           const float* __restrict__ fe,
                           const float* __restrict__ v_emb,
                           const float* __restrict__ size_emb,
                           float* __restrict__ comb)
{
    int n = blockIdx.x, t = threadIdx.x;
    comb[(long)n * 768 + t]        = v_emb[(long)v[n] * HD + t];
    comb[(long)n * 768 + 256 + t]  = fe[(long)n * HD + t];
    comb[(long)n * 768 + 512 + t]  = size_emb[(long)sz[n] * HD + t];
}

// ---- ef[e] = [ h[row] | h[col] | radial | radial | pad2 ] ; radial/coord_diff ----
__global__ void edge_build(const float* __restrict__ h, const float* __restrict__ x,
                           const int* __restrict__ rows, const int* __restrict__ cols,
                           float* __restrict__ ef, float* __restrict__ radial,
                           float* __restrict__ cdiff)
{
    int e = blockIdx.x, t = threadIdx.x;
    int r = rows[e], c = cols[e];
    ef[(long)e * EFKS + t]      = h[(long)r * HD + t];
    ef[(long)e * EFKS + HD + t] = h[(long)c * HD + t];
    if (t == 0) {
        float d0 = x[r*3+0] - x[c*3+0];
        float d1 = x[r*3+1] - x[c*3+1];
        float d2 = x[r*3+2] - x[c*3+2];
        float rad = d0*d0 + d1*d1 + d2*d2;
        float inv = 1.0f / (sqrtf(rad) + 1.0f);
        radial[e]    = rad;
        cdiff[e*3+0] = d0 * inv;
        cdiff[e*3+1] = d1 * inv;
        cdiff[e*3+2] = d2 * inv;
        ef[(long)e * EFKS + 512] = rad;
        ef[(long)e * EFKS + 513] = rad;
    }
}

// ---- msg *= sigmoid(msg . Watt + batt) ----
__global__ void att_scale(float* __restrict__ msg, const float* __restrict__ Watt,
                          const float* __restrict__ batt)
{
    __shared__ float red[256];
    int e = blockIdx.x, t = threadIdx.x;
    red[t] = msg[(long)e * HD + t] * Watt[t];
    __syncthreads();
    for (int s = 128; s > 0; s >>= 1) { if (t < s) red[t] += red[t + s]; __syncthreads(); }
    float a = 1.0f / (1.0f + __expf(-(red[0] + batt[0])));
    msg[(long)e * HD + t] *= a;
}

// ---- agg[row[e]] += msg[e] (segment_sum) ----
__global__ void scatter_add(const float* __restrict__ msg, const int* __restrict__ rows,
                            float* __restrict__ agg)
{
    int e = blockIdx.x, t = threadIdx.x;
    atomicAdd(&agg[(long)rows[e] * HD + t], msg[(long)e * HD + t]);
}

// ---- hcat[n] = [h[n] | agg[n]] ----
__global__ void build_hcat(const float* __restrict__ h, const float* __restrict__ agg,
                           float* __restrict__ hcat)
{
    int n = blockIdx.x, t = threadIdx.x;
    hcat[(long)n * 512 + t]      = h[(long)n * HD + t];
    hcat[(long)n * 512 + HD + t] = agg[(long)n * HD + t];
}

// ---- tt = tanh(t1[e] . Wc2) * 30 ; x[row] += cdiff[e] * tt ----
__global__ void coord_update(const float* __restrict__ t1, const float* __restrict__ Wc2,
                             const int* __restrict__ rows, const float* __restrict__ cdiff,
                             float* __restrict__ x)
{
    __shared__ float red[256];
    int e = blockIdx.x, t = threadIdx.x;
    red[t] = t1[(long)e * HD + t] * Wc2[t];
    __syncthreads();
    for (int s = 128; s > 0; s >>= 1) { if (t < s) red[t] += red[t + s]; __syncthreads(); }
    float tt = tanhf(red[0]) * CRANGE;
    if (t < 3) atomicAdd(&x[(long)rows[e] * 3 + t], cdiff[e * 3 + t] * tt);
}

// ---- x *= mask (per node) ----
__global__ void mask_x(float* __restrict__ x, const float* __restrict__ mask)
{
    int i = blockIdx.x * blockDim.x + threadIdx.x;
    if (i < NN * 3) x[i] *= mask[i / 3];
}

// ---- output head ----
__global__ void head1(const float* __restrict__ h, const int* __restrict__ pidx,
                      const float* __restrict__ val, const float* __restrict__ oW1,
                      const float* __restrict__ ob1, float* __restrict__ t1)
{
    int b = blockIdx.x, j = threadIdx.x;
    long row = (long)b * NNODES + pidx[b];
    float acc = ob1[j];
    for (int k = 0; k < HD; ++k) acc += h[row * HD + k] * oW1[(long)k * HD + j];
    acc += val[b] * oW1[(long)HD * HD + j];   // oW1 is [H+1, H]
    t1[b * HD + j] = siluf(acc);
}

__global__ void head2(const float* __restrict__ t1, const float* __restrict__ oW2,
                      const float* __restrict__ ob2, float* __restrict__ out)
{
    int b = blockIdx.y;
    int vtx = blockIdx.x * 256 + threadIdx.x;
    if (vtx >= VOC) return;
    float acc = ob2[vtx];
    for (int k = 0; k < HD; ++k) acc += t1[b * HD + k] * oW2[(long)k * VOC + vtx];
    out[(long)b * VOC + vtx] = acc;
}

// =====================================================================
extern "C" void kernel_launch(void* const* d_in, const int* in_sizes, int n_in,
                              void* d_out, int out_size, void* d_ws, size_t ws_size,
                              hipStream_t stream)
{
    (void)in_sizes; (void)n_in; (void)out_size; (void)ws_size;

    const float* feature  = (const float*)d_in[0];
    const int*   v        = (const int*)  d_in[1];
    const int*   sz       = (const int*)  d_in[2];
    const float* pos      = (const float*)d_in[3];
    const int*   edges    = (const int*)  d_in[4];
    const float* mask     = (const float*)d_in[5];
    const int*   pidx     = (const int*)  d_in[6];
    const float* val      = (const float*)d_in[7];
    const float* v_emb    = (const float*)d_in[8];
    const float* size_emb = (const float*)d_in[9];
    const float* fW1 = (const float*)d_in[10]; const float* fb1 = (const float*)d_in[11];
    const float* fW2 = (const float*)d_in[12]; const float* fb2 = (const float*)d_in[13];
    const float* pW1 = (const float*)d_in[14]; const float* pb1 = (const float*)d_in[15];
    const float* pW2 = (const float*)d_in[16]; const float* pb2 = (const float*)d_in[17];
    const float* pW3 = (const float*)d_in[18]; const float* pb3 = (const float*)d_in[19];
    const float* We1 = (const float*)d_in[20]; const float* be1 = (const float*)d_in[21];
    const float* We2 = (const float*)d_in[22]; const float* be2 = (const float*)d_in[23];
    const float* Watt= (const float*)d_in[24]; const float* batt= (const float*)d_in[25];
    const float* Wn1 = (const float*)d_in[26]; const float* bn1 = (const float*)d_in[27];
    const float* Wn2 = (const float*)d_in[28]; const float* bn2 = (const float*)d_in[29];
    const float* Wc1 = (const float*)d_in[30]; const float* bc1 = (const float*)d_in[31];
    const float* Wc2 = (const float*)d_in[32];
    const float* oW1 = (const float*)d_in[33]; const float* ob1 = (const float*)d_in[34];
    const float* oW2 = (const float*)d_in[35]; const float* ob2 = (const float*)d_in[36];

    // ---- workspace layout (reused across phases) ----
    char* ws = (char*)d_ws;
    size_t off = 0;
    auto alloc = [&](size_t bytes) -> void* {
        void* p = (void*)(ws + off);
        off += (bytes + 255) & ~(size_t)255;
        return p;
    };
    float* h      = (float*)alloc((size_t)NN * HD * 4);    // persistent node features
    float* x      = (float*)alloc((size_t)NN * 3 * 4);     // persistent coords
    float* big0   = (float*)alloc((size_t)NN * 768 * 4);   // comb / ef (EE*516 fits)
    float* big1   = (float*)alloc((size_t)NN * 768 * 4);   // hp1 / hcat
    float* t0     = (float*)alloc((size_t)NN * HD * 4);    // fe1 / msg1 / t1(coord)
    float* t1b    = (float*)alloc((size_t)NN * HD * 4);    // fe  / msg
    float* t2     = (float*)alloc((size_t)NN * HD * 4);    // hp2 / agg / nh1
    float* radial = (float*)alloc((size_t)EE * 4);
    float* cdiff  = (float*)alloc((size_t)EE * 3 * 4);
    float* ht1    = (float*)alloc((size_t)BSZ * HD * 4);
    // transposed bf16 weights [N][KP], zero-padded K
    __bf16* fW1t = (__bf16*)alloc((size_t)HD * 512 * 2);
    __bf16* fW2t = (__bf16*)alloc((size_t)HD * 256 * 2);
    __bf16* pW1t = (__bf16*)alloc((size_t)768 * 768 * 2);
    __bf16* pW2t = (__bf16*)alloc((size_t)HD * 768 * 2);
    __bf16* pW3t = (__bf16*)alloc((size_t)HD * 256 * 2);
    __bf16* We1t = (__bf16*)alloc((size_t)9 * HD * EFKP * 2);
    __bf16* We2t = (__bf16*)alloc((size_t)9 * HD * 256 * 2);
    __bf16* Wn1t = (__bf16*)alloc((size_t)9 * HD * 512 * 2);
    __bf16* Wn2t = (__bf16*)alloc((size_t)9 * HD * 256 * 2);
    __bf16* Wc1t = (__bf16*)alloc((size_t)9 * HD * 256 * 2);

    dim3 blk(256);
    auto wt = [&](const float* W, __bf16* Wt, int K, int N, int KP, int L) {
        dim3 g((KP + 255) / 256, N, L);
        wtrans<<<g, blk, 0, stream>>>(W, Wt, K, N, KP);
    };
    auto gemm = [&](const float* A, const __bf16* Bt, const float* bias,
                    const float* res, const float* mrow, float* C,
                    int M, int N, int K, int lda, int ldb, int ldc, int act) {
        dim3 grid((N + BN - 1) / BN, (M + BM - 1) / BM);
        gemm_bf16_wmma<<<grid, blk, 0, stream>>>(A, Bt, bias, res, mrow, C,
                                                 M, N, K, lda, ldb, ldc, act);
    };

    // ---- one-time (per call) weight convert+transpose ----
    wt(fW1, fW1t, 512, HD,  512,  1);
    wt(fW2, fW2t, HD,  HD,  256,  1);
    wt(pW1, pW1t, 768, 768, 768,  1);
    wt(pW2, pW2t, 768, HD,  768,  1);
    wt(pW3, pW3t, HD,  HD,  256,  1);
    wt(We1, We1t, EFK, HD,  EFKP, 9);
    wt(We2, We2t, HD,  HD,  256,  9);
    wt(Wn1, Wn1t, 512, HD,  512,  9);
    wt(Wn2, Wn2t, HD,  HD,  256,  9);
    wt(Wc1, Wc1t, HD,  HD,  256,  9);

    // ---- node embedding ----
    gemm(feature, fW1t, fb1, nullptr, nullptr, t0,  NN, HD, FEATD, FEATD, 512, HD, 1); // fe1
    gemm(t0,      fW2t, fb2, nullptr, nullptr, t1b, NN, HD, HD,    HD,    256, HD, 0); // fe
    build_comb<<<NN, 256, 0, stream>>>(v, sz, t1b, v_emb, size_emb, big0);
    gemm(big0, pW1t, pb1, nullptr, nullptr, big1, NN, 768, 768, 768, 768, 768, 1);     // hp1
    gemm(big1, pW2t, pb2, nullptr, nullptr, t2,   NN, HD,  768, 768, 768, HD,  1);     // hp2
    gemm(t2,   pW3t, pb3, nullptr, mask,    h,    NN, HD,  HD,  HD,  256, HD,  0);     // h=(.)*m
    hipMemcpyAsync(x, pos, (size_t)NN * 3 * 4, hipMemcpyDeviceToDevice, stream);

    // ---- 9 E_GCL layers over 3 phases (fwd / reversed / fwd edges) ----
    for (int p = 0; p < 3; ++p) {
        const int* rows = (p == 1) ? (edges + EE) : edges;
        const int* cols = (p == 1) ? edges        : (edges + EE);
        for (int i = 0; i < 3; ++i) {
            int l = p * 3 + i;
            edge_build<<<EE, 256, 0, stream>>>(h, x, rows, cols, big0, radial, cdiff);
            gemm(big0, We1t + (long)l*HD*EFKP, be1 + l*HD, nullptr, nullptr, t0,
                 EE, HD, EFK, EFKS, EFKP, HD, 1);                                      // msg1
            gemm(t0,   We2t + (long)l*HD*256,  be2 + l*HD, nullptr, nullptr, t1b,
                 EE, HD, HD, HD, 256, HD, 1);                                          // msg
            att_scale<<<EE, 256, 0, stream>>>(t1b, Watt + (long)l*HD, batt + l);
            hipMemsetAsync(t2, 0, (size_t)NN * HD * 4, stream);
            scatter_add<<<EE, 256, 0, stream>>>(t1b, rows, t2);                        // agg
            build_hcat<<<NN, 256, 0, stream>>>(h, t2, big1);                           // [h|agg]
            gemm(big1, Wn1t + (long)l*HD*512, bn1 + l*HD, nullptr, nullptr, t2,
                 NN, HD, 2*HD, 2*HD, 512, HD, 1);                                      // nh1
            gemm(t2,   Wn2t + (long)l*HD*256, bn2 + l*HD, h, mask, h,
                 NN, HD, HD, HD, 256, HD, 0);                                          // h=(h+nh)*m
            gemm(t1b,  Wc1t + (long)l*HD*256, bc1 + l*HD, nullptr, nullptr, t0,
                 EE, HD, HD, HD, 256, HD, 1);                                          // coord t
            coord_update<<<EE, 256, 0, stream>>>(t0, Wc2 + (long)l*HD, rows, cdiff, x);
            mask_x<<<(NN*3 + 255)/256, 256, 0, stream>>>(x, mask);
        }
    }

    // ---- output head ----
    head1<<<BSZ, 256, 0, stream>>>(h, pidx, val, oW1, ob1, ht1);
    dim3 g2((VOC + 255) / 256, BSZ);
    head2<<<g2, 256, 0, stream>>>(ht1, oW2, ob2, (float*)d_out);
}
